// NMSLayer_2010044695233
// MI455X (gfx1250) — 8-bit matrix operand order repaired and checked by reading, compiled, not run on hardware
//
#include <hip/hip_runtime.h>
#include <hip/hip_bf16.h>

#define BS   32
#define NROW 10647
#define NC   85
#define KTOP 2048
#define SORTN 16384
#define CONF_THRESH 0.9f
#define NMS_THRESH  0.3f

#define OFF_SCORES 0ull
#define OFF_CLS    1362816ull
#define OFF_KEYS   2725632ull
#define OFF_TOPK   5451264ull
#define OFF_SEL    5975552ull
#define OFF_SUP    8072704ull

typedef int v8i __attribute__((ext_vector_type(8)));

__device__ __forceinline__ unsigned sortable_f32(float f) {
    unsigned u = __float_as_uint(f);
    return (u & 0x80000000u) ? ~u : (u | 0x80000000u);
}

__device__ __forceinline__ int onehot_word(int c, int base) {
    unsigned d = (unsigned)(c - base);
    return (d < 4u) ? (int)(1u << (8u * d)) : 0;
}

__global__ void k_score(const float* __restrict__ x,
                        float* __restrict__ scores,
                        int*   __restrict__ clsidx,
                        unsigned long long* __restrict__ keys) {
    int w    = (blockIdx.x * blockDim.x + threadIdx.x) >> 5;
    int lane = threadIdx.x & 31;
    const float* p = x + (size_t)w * NC;
    float obj = p[4];
    float v  = p[5 + lane];              int ki = lane;
    float v1 = p[5 + lane + 32];         if (v1 > v) { v = v1; ki = lane + 32; }
    if (lane < 16) { float v2 = p[5 + lane + 64]; if (v2 > v) { v = v2; ki = lane + 64; } }
    #pragma unroll
    for (int off = 16; off; off >>= 1) {
        float ov = __shfl_xor(v, off, 32);
        int   oi = __shfl_xor(ki, off, 32);
        if (ov > v || (ov == v && oi < ki)) { v = ov; ki = oi; }
    }
    if (lane == 0) {
        scores[w] = v;
        clsidx[w] = ki;
        float key = (obj > CONF_THRESH) ? v : -INFINITY;
        unsigned hi = sortable_f32(key);
        unsigned i  = (unsigned)(w % NROW);
        keys[w] = ((unsigned long long)hi << 32) | (unsigned)(~i);
    }
}

__global__ void k_topk(const unsigned long long* __restrict__ keys,
                       unsigned long long* __restrict__ topk) {
    extern __shared__ unsigned long long arr[];
    const int b = blockIdx.x, t = threadIdx.x;
    const unsigned long long* kb = keys + (size_t)b * NROW;
    for (int i = t; i < SORTN; i += 1024) {
        if (i + 1024 < NROW) __builtin_prefetch(&kb[i + 1024], 0, 1);
        arr[i] = (i < NROW) ? kb[i] : 0ull;
    }
    __syncthreads();
    for (int k = 2; k <= SORTN; k <<= 1) {
        for (int j = k >> 1; j > 0; j >>= 1) {
            for (int i = t; i < SORTN; i += 1024) {
                int ixj = i ^ j;
                if (ixj > i) {
                    bool desc = ((i & k) == 0);
                    unsigned long long a0 = arr[i], a1 = arr[ixj];
                    bool sw = desc ? (a0 < a1) : (a0 > a1);
                    if (sw) { arr[i] = a1; arr[ixj] = a0; }
                }
            }
            __syncthreads();
        }
    }
    for (int i = t; i < KTOP; i += 1024)
        topk[(size_t)b * KTOP + i] = arr[i];
}

__global__ void k_gather(const float* __restrict__ x,
                         const float* __restrict__ scores,
                         const int*   __restrict__ clsidx,
                         const unsigned long long* __restrict__ topk,
                         float* __restrict__ sel) {
    int g = blockIdx.x * blockDim.x + threadIdx.x;
    int b = g / KTOP;
    unsigned long long key = topk[g];
    unsigned idx = ~(unsigned)(key & 0xFFFFFFFFull);
    bool inr = idx < (unsigned)NROW;
    unsigned ridx = inr ? idx : 0u;
    size_t row = (size_t)b * NROW + ridx;
    const float* p = x + row * NC;
    float xc = p[0], yc = p[1], wv = p[2], hv = p[3], obj = p[4];
    bool valid = inr && (obj > CONF_THRESH);
    float* s = sel + (size_t)g * 8;
    s[0] = xc - wv * 0.5f;  s[1] = yc - hv * 0.5f;
    s[2] = xc + wv * 0.5f;  s[3] = yc + hv * 0.5f;
    s[4] = obj;
    s[5] = scores[row];
    s[6] = (float)clsidx[row];
    s[7] = valid ? 1.0f : 0.0f;
}

__global__ void k_suppress(const float* __restrict__ sel,
                           unsigned short* __restrict__ sup) {
    int wid  = (blockIdx.x * blockDim.x + threadIdx.x) >> 5;
    int lane = threadIdx.x & 31;
    int img  = wid >> 14;
    int tile = wid & 16383;
    int ti = tile >> 7, tj = tile & 127;
    if (ti > tj) return;
    int I0 = ti << 4,  J0 = tj << 4;
    const float*  selb = sel + (size_t)img * KTOP * 8;
    const float4* sel4 = (const float4*)selb;

    int nn = lane & 15, half = lane >> 4;
    int cls_i = (int)selb[(size_t)(I0 + nn) * 8 + 6];
    int cls_j = (int)selb[(size_t)(J0 + nn) * 8 + 6];

    v8i A0{}, A1{}, B0{}, B1{};
    #pragma unroll
    for (int v = 0; v < 8; ++v) {
        int ab = 4 * v + 32 * half;
        int bb = 4 * v + 32 * half;
        A0[v] = onehot_word(cls_i, ab);       A1[v] = onehot_word(cls_i, ab + 64);
        B0[v] = onehot_word(cls_j, bb);       B1[v] = onehot_word(cls_j, bb + 64);
    }
    v8i acc{};
    acc = __builtin_amdgcn_wmma_i32_16x16x64_iu8(false, A0, false, B0, acc, false, false);
    acc = __builtin_amdgcn_wmma_i32_16x16x64_iu8(false, A1, false, B1, acc, false, false);

    float4 bj = sel4[(size_t)(J0 + nn) * 2];
    float areaj = (bj.z - bj.x) * (bj.w - bj.y);
    #pragma unroll
    for (int r = 0; r < 8; ++r) {
        int m = r + 8 * half;
        float4 bi = sel4[(size_t)(I0 + m) * 2];
        float areai = (bi.z - bi.x) * (bi.w - bi.y);
        float xx1 = fmaxf(bi.x, bj.x), yy1 = fmaxf(bi.y, bj.y);
        float xx2 = fminf(bi.z, bj.z), yy2 = fminf(bi.w, bj.w);
        float inter = fmaxf(xx2 - xx1, 0.0f) * fmaxf(yy2 - yy1, 0.0f);
        float iou = inter / fmaxf(areai + areaj - inter, 1e-9f);
        bool pr = (acc[r] > 0) && (iou >= NMS_THRESH);
        unsigned mask = (unsigned)__ballot(pr);
        if (lane == 0)
            sup[((size_t)img * KTOP + (I0 + r)) * 128 + tj] = (unsigned short)(mask & 0xFFFFu);
        if (lane == 16)
            sup[((size_t)img * KTOP + (I0 + r + 8)) * 128 + tj] = (unsigned short)(mask >> 16);
    }
}

__global__ void k_greedy_out(const float* __restrict__ sel,
                             const unsigned* __restrict__ sup,
                             float* __restrict__ out) {
    extern __shared__ unsigned rows[];
    __shared__ unsigned keepw[KTOP / 32];
    const int b = blockIdx.x, t = threadIdx.x;
    const float* selb = sel + (size_t)b * KTOP * 8;
    const unsigned* supb = sup + (size_t)b * KTOP * 64;

    if (t < 64) {
        unsigned w = 0;
        for (int j = 0; j < 32; ++j)
            if (selb[(size_t)(t * 32 + j) * 8 + 7] != 0.0f) w |= 1u << j;
        keepw[t] = w;
    }

    for (int c = 0; c < 8; ++c) {
        __syncthreads();
        const uint4* src = (const uint4*)(supb + (size_t)c * 256 * 64);
        uint4* dst = (uint4*)rows;
        for (int q = t; q < 256 * 16; q += 256) {
            if (q + 256 < 256 * 16) __builtin_prefetch(&src[q + 256], 0, 1);
            dst[q] = src[q];
        }
        __syncthreads();
        for (int ii = 0; ii < 256; ++ii) {
            int i = (c << 8) + ii;
            bool ki = (keepw[i >> 5] >> (i & 31)) & 1u;
            __syncthreads();
            if (ki && t < 64) {
                int iw = i >> 5;
                unsigned s = rows[ii * 64 + t];
                if (t < iw)       s = 0u;
                else if (t == iw) s &= ~((2u << (i & 31)) - 1u);
                keepw[t] &= ~s;
            }
            __syncthreads();
        }
    }

    for (int k = t; k < KTOP; k += 256) {
        float kf = ((keepw[k >> 5] >> (k & 31)) & 1u) ? 1.0f : 0.0f;
        const float* s = selb + (size_t)k * 8;
        float4* o = (float4*)(out + ((size_t)b * KTOP + k) * 8);
        float4 v0, v1;
        v0.x = (float)b * kf; v0.y = s[0] * kf; v0.z = s[1] * kf; v0.w = s[2] * kf;
        v1.x = s[3] * kf;     v1.y = s[4] * kf; v1.z = s[5] * kf; v1.w = s[6] * kf;
        o[0] = v0;  o[1] = v1;
        out[(size_t)BS * KTOP * 8 + (size_t)b * KTOP + k] = kf;
    }
}

extern "C" void kernel_launch(void* const* d_in, const int* in_sizes, int n_in,
                              void* d_out, int out_size, void* d_ws, size_t ws_size,
                              hipStream_t stream) {
    const float* x = (const float*)d_in[0];
    float* out = (float*)d_out;
    char* ws = (char*)d_ws;

    float*              scores = (float*)(ws + OFF_SCORES);
    int*                clsi   = (int*)(ws + OFF_CLS);
    unsigned long long* keys   = (unsigned long long*)(ws + OFF_KEYS);
    unsigned long long* topk   = (unsigned long long*)(ws + OFF_TOPK);
    float*              sel    = (float*)(ws + OFF_SEL);
    unsigned short*     sup16  = (unsigned short*)(ws + OFF_SUP);
    unsigned*           sup32  = (unsigned*)(ws + OFF_SUP);

    k_score<<<(BS * NROW) / 8, 256, 0, stream>>>(x, scores, clsi, keys);
    k_topk<<<BS, 1024, SORTN * sizeof(unsigned long long), stream>>>(keys, topk);
    k_gather<<<(BS * KTOP) / 256, 256, 0, stream>>>(x, scores, clsi, topk, sel);
    k_suppress<<<(BS * 128 * 128) / 8, 256, 0, stream>>>(sel, sup16);
    k_greedy_out<<<BS, 256, 65536, stream>>>(sel, sup32, out);
}
